// CLIPVisionTower_Nuwa_Next_7610682049079
// MI455X (gfx1250) — compile-verified, matrix operand order checked by reading, not run on hardware
//
#include <hip/hip_runtime.h>
#include <hip/hip_bf16.h>
#include <math.h>

// ---------------- static config (matches reference) ----------------
#define BATCH 64
#define NTOK  576          // 24*24
#define DIM   1024
#define KKEEP 144
#define GRIDW 24
#define INV_THR 0.05976143046671968f   // 1/sqrt(280)

typedef __attribute__((ext_vector_type(16))) __bf16 v16bf;
typedef __attribute__((ext_vector_type(2)))  __bf16 v2bf;
typedef __attribute__((ext_vector_type(8)))  float  v8f;

// fast hardware sqrt (penalty distances are small & well-conditioned)
static __device__ __forceinline__ float fsqrt(float x) {
#if __has_builtin(__builtin_amdgcn_sqrtf)
    return __builtin_amdgcn_sqrtf(x);
#else
    return sqrtf(x);
#endif
}

// float pair -> packed bf16 (round-half-up) in minimal VALU ops
static __device__ __forceinline__ v2bf f2bf2(float x, float y) {
#if __has_builtin(__builtin_amdgcn_cvt_pk_bf16_f32)
    return __builtin_amdgcn_cvt_pk_bf16_f32(x, y);
#else
    unsigned ux = __builtin_bit_cast(unsigned, x) + 0x8000u;
    unsigned uy = __builtin_bit_cast(unsigned, y) + 0x8000u;
#if __has_builtin(__builtin_amdgcn_perm)
    // dst = {uy[3],uy[2],ux[3],ux[2]} : both high halves packed in ONE v_perm_b32
    unsigned pk = __builtin_amdgcn_perm(uy, ux, 0x07060302u);
#else
    unsigned pk = (ux >> 16) | (uy & 0xFFFF0000u);
#endif
    return __builtin_bit_cast(v2bf, pk);
#endif
}

// K-slot mapping for 16-bit A/B fragments (ISA 7.12.2, 16-bit 16x32 layout):
// VGPR r in lane-half 'hi' holds K = kslot(r,hi), kslot+1
static __device__ __forceinline__ int kslot(int r, int hi) {
    return 2 * r + (r >= 4 ? 8 : 0) + (hi ? 8 : 0);
}

// swizzle within a 32-wide K block: store order [0..7, 16..23, 8..15, 24..31]
// (swap bits 3 and 4) so each lane-half's fragment is 16 contiguous bf16.
static __device__ __forceinline__ int kswz(int k) {
    return (k & ~24) | ((k & 8) << 1) | ((k & 16) >> 1);
}

static __device__ __forceinline__ v8f wmma_bf16(v16bf a, v16bf b, v8f c) {
    // D = A(16x32) * B(32x16) + C, f32 accumulate
    return __builtin_amdgcn_wmma_f32_16x16x32_bf16(
        false, a, false, b, (short)0, c, false, false);
}

// A fragment from swizzled LDS: one contiguous 32-byte read (2x ds_load_b128)
static __device__ __forceinline__ v16bf lds_afrag_sw(const __bf16* base, int stride, int lane) {
    int m = lane & 15, hi = (lane >> 4) & 1;
    return *(const v16bf*)(base + m * stride + hi * 16);
}

// G1 B fragment: per-lane token row, d-pairs are contiguous runs -> 4x float4 loads
static __device__ __forceinline__ v16bf g1_bfrag(const float* xb, int hi) {
    union { v16bf v; v2bf h[8]; } u;
    const float* p = xb + hi * 8;
#pragma unroll
    for (int q = 0; q < 4; ++q) {
        const int off = (q & 1) * 4 + (q >> 1) * 16;   // 0,4,16,20
        float4 f = *(const float4*)(p + off);
        u.h[2 * q]     = f2bf2(f.x, f.y);
        u.h[2 * q + 1] = f2bf2(f.z, f.w);
    }
    return u.v;
}

// G2 B fragment: per-lane d column, 32 token rows (stride DIM)
static __device__ __forceinline__ v16bf g2_bfrag(const float* xc, int hi) {
    union { v16bf v; v2bf h[8]; } u;
#pragma unroll
    for (int r = 0; r < 8; ++r) {
        int k0 = kslot(r, hi);
        u.h[r] = f2bf2(xc[(size_t)k0 * DIM], xc[(size_t)(k0 + 1) * DIM]);
    }
    return u.v;
}

// ---------------- kernel 1: inverse row norms of hidden_sim ----------------
__global__ __launch_bounds__(256) void row_norms(const float* __restrict__ x,
                                                 float* __restrict__ rinv) {
    int row  = blockIdx.x * 8 + (threadIdx.x >> 5);   // 8 wave32 rows per block
    int lane = threadIdx.x & 31;
    const float4* p4 = (const float4*)(x + (size_t)row * DIM);
    float s = 0.f;
#pragma unroll 4
    for (int i = lane; i < DIM / 4; i += 32) {
        float4 v = p4[i];
        s += v.x * v.x + v.y * v.y + v.z * v.z + v.w * v.w;
    }
#pragma unroll
    for (int off = 16; off; off >>= 1) s += __shfl_xor(s, off, 32);
    if (lane == 0) rinv[row] = 1.0f / fmaxf(fsqrt(s), 1e-12f);
}

// ---------------- kernel 2: benchmark token selection ----------------
__global__ __launch_bounds__(288) void bench_select(const float* __restrict__ metric,
                                                    int* __restrict__ bench,
                                                    int* __restrict__ high,
                                                    int* __restrict__ out_idx) {
    const int b = blockIdx.x, tid = threadIdx.x;   // blockDim == 288
    __shared__ float sm[NTOK];
    __shared__ int   ctok[288];
    __shared__ float cs[288];
    __shared__ int   keep[288];
    __shared__ int   btok[KKEEP];
    __shared__ float bs[KKEEP];
    __shared__ float srt[KKEEP];

    for (int i = tid; i < NTOK; i += 288) sm[i] = metric[b * NTOK + i];
    __syncthreads();

    // per-2x2-region top-2 (lax.top_k tie rule: strict > keeps lower index)
    if (tid < 144) {
        int r = tid, ry = r / 12, rx = r % 12;
        float v[4]; int g[4];
#pragma unroll
        for (int j = 0; j < 4; ++j) {
            int gg = (ry * 2 + (j >> 1)) * GRIDW + rx * 2 + (j & 1);
            g[j] = gg; v[j] = sm[gg];
        }
        int i1 = 0;
        for (int j = 1; j < 4; ++j) if (v[j] > v[i1]) i1 = j;
        int i2 = -1;
        for (int j = 0; j < 4; ++j) { if (j == i1) continue; if (i2 < 0 || v[j] > v[i2]) i2 = j; }
        ctok[2 * r]     = g[i1]; cs[2 * r]     = v[i1];
        ctok[2 * r + 1] = g[i2]; cs[2 * r + 1] = v[i2];
    }
    __syncthreads();

    // global top-K among 288 candidates via exact rank (top_k tie rule)
    {
        float st = cs[tid]; int rank = 0;
        for (int j = 0; j < 288; ++j) {
            float sj = cs[j];
            rank += (sj > st) || (sj == st && j < tid);
        }
        keep[tid] = (rank < KKEEP);
    }
    __syncthreads();

    // compact into ascending token order (tokens are unique)
    if (keep[tid]) {
        int tk = ctok[tid], pos = 0;
        for (int j = 0; j < 288; ++j) pos += (keep[j] && ctok[j] < tk);
        btok[pos] = tk; bs[pos] = cs[tid];
    }
    __syncthreads();

    // sort bscores ascending for the linear quantile
    if (tid < KKEEP) {
        float vv = bs[tid]; int pos = 0;
        for (int j = 0; j < KKEEP; ++j) {
            float vj = bs[j];
            pos += (vj < vv) || (vj == vv && j < tid);
        }
        srt[pos] = vv;
    }
    __syncthreads();

    if (tid < KKEEP) {
        float h  = 0.55f * (float)(KKEEP - 1);      // 78.65
        int   lo = (int)h;
        float thr = srt[lo] + (srt[lo + 1] - srt[lo]) * (h - (float)lo);
        bench[b * KKEEP + tid]   = btok[tid];
        high[b * KKEEP + tid]    = (bs[tid] >= thr) ? 1 : 0;
        out_idx[b * KKEEP + tid] = btok[tid];
    }
}

// ---------------- kernel 3: fused sim-row GEMM + mask/normalize + aggregate ----------------
__global__ __launch_bounds__(256) void nuwa_main(const float* __restrict__ xsim,
                                                 const float* __restrict__ xagg,
                                                 const float* __restrict__ rinvg,
                                                 const int*   __restrict__ bench,
                                                 const int*   __restrict__ high,
                                                 float*       __restrict__ out) {
    const int kt   = blockIdx.x;           // 0..8  (16 benchmark rows each)
    const int b    = blockIdx.y;           // 0..63
    const int tid  = threadIdx.x;
    const int lane = tid & 31, wave = tid >> 5;
    const int hi   = (lane >> 4) & 1, nl = lane & 15;

    __shared__ __align__(32) __bf16 sA[16 * DIM];     // 32 KB bench rows (swizzled bf16); reused for W
    __shared__ float sW[16 * 577];                    // padded to dodge bank conflicts
    __shared__ float sRowSum[16];
    __shared__ float sRinv[16];
    __shared__ float sRn[NTOK];
    __shared__ int   sIdx[16];
    __shared__ int   sHigh[16];
    __shared__ int   sTy[16], sTx[16];

    const size_t baseSim = (size_t)b * NTOK * DIM;
    if (tid < 16) {
        int t = bench[b * KKEEP + kt * 16 + tid];
        sIdx[tid]    = t;
        sTy[tid]     = t / GRIDW;
        sTx[tid]     = t % GRIDW;
        sHigh[tid]   = high[b * KKEEP + kt * 16 + tid];
        sRinv[tid]   = rinvg[b * NTOK + t];
        sRowSum[tid] = 0.f;
    }
    for (int i = tid; i < NTOK; i += 256) sRn[i] = rinvg[b * NTOK + i];
    __syncthreads();
    // stage A rows as swizzled bf16 (pairwise packed convert)
    for (int i = tid; i < 16 * (DIM / 2); i += 256) {
        int m = i >> 9, d = (i & 511) * 2;
        float2 f = *(const float2*)(xsim + baseSim + (size_t)sIdx[m] * DIM + d);
        int pos = (d & ~31) | kswz(d & 31);
        *(v2bf*)(sA + m * DIM + pos) = f2bf2(f.x, f.y);
    }
    __syncthreads();

    // ---- G1: W[16 x 576] = A(bench rows) x xsim^T, fused scale/penalty/mask ----
    float rs[8] = {0.f, 0.f, 0.f, 0.f, 0.f, 0.f, 0.f, 0.f};
    for (int p = wave; p < NTOK / 32; p += 8) {          // two 16-wide tiles per pass
        int n0 = p * 32;
        const float* xb0 = xsim + baseSim + (size_t)(n0 + nl) * DIM;
        const float* xb1 = xb0 + (size_t)16 * DIM;
        v8f acc0 = {}, acc1 = {};
#pragma unroll 2
        for (int kc = 0; kc < DIM / 32; ++kc) {
            v16bf af = lds_afrag_sw(sA + kc * 32, DIM, lane);
            v16bf b0 = g1_bfrag(xb0 + kc * 32, hi);
            v16bf b1 = g1_bfrag(xb1 + kc * 32, hi);
            acc0 = wmma_bf16(af, b0, acc0);
            acc1 = wmma_bf16(af, b1, acc1);
        }
#pragma unroll
        for (int t2 = 0; t2 < 2; ++t2) {
            v8f acc = t2 ? acc1 : acc0;
            int nn = n0 + 16 * t2 + nl;
            float rinvN = sRn[nn];
            int ny = nn / GRIDW, nx = nn % GRIDW;
#pragma unroll
            for (int r = 0; r < 8; ++r) {
                int m  = r + 8 * hi;                   // C/D layout: VGPR r -> M=r / M=r+8
                float w = fmaxf(acc[r] * sRinv[m] * rinvN, 0.f);
                float dy = (float)(sTy[m] - ny), dx = (float)(sTx[m] - nx);
                float pen = 1.f - fminf(fsqrt(dy * dy + dx * dx) * INV_THR, 1.f);
                w *= pen;
                if (sHigh[m] && nn != sIdx[m]) w = 0.f;
                sW[m * 577 + nn] = w;
                rs[r] += w;
            }
        }
    }
#pragma unroll
    for (int r = 0; r < 8; ++r) atomicAdd(&sRowSum[r + 8 * hi], rs[r]);   // ds_add_f32
    __syncthreads();

    // ---- normalize rows, scatter self-weight = 1, requantize to swizzled bf16 ----
    __bf16* sWb = sA;                                  // reuse A staging (16*576 bf16)
    for (int i = tid; i < 16 * (NTOK / 2); i += 256) {
        int m = i / (NTOK / 2), j = (i - m * (NTOK / 2)) * 2;
        float inv = 1.0f / (sRowSum[m] + 1e-8f);
        float w0 = sW[m * 577 + j] * inv;
        float w1 = sW[m * 577 + j + 1] * inv;
        if (j     == sIdx[m]) w0 = 1.f;
        if (j + 1 == sIdx[m]) w1 = 1.f;
        int pos = (j & ~31) | kswz(j & 31);
        *(v2bf*)(sWb + m * NTOK + pos) = f2bf2(w0, w1);
    }
    __syncthreads();

    // ---- G2: out[16 x 1024] = W[16 x 576] x xagg[576 x 1024] ----
    const size_t baseAgg = (size_t)b * NTOK * DIM;
    for (int p = wave; p < DIM / 32; p += 8) {           // two 16-wide tiles per pass
        int d0 = p * 32;
        const float* xc0 = xagg + baseAgg + d0 + nl;
        const float* xc1 = xc0 + 16;
        v8f acc0 = {}, acc1 = {};
#pragma unroll 2
        for (int kb = 0; kb < NTOK / 32; ++kb) {
            v16bf af = lds_afrag_sw(sWb + kb * 32, NTOK, lane);
            v16bf b0 = g2_bfrag(xc0 + (size_t)kb * 32 * DIM, hi);
            v16bf b1 = g2_bfrag(xc1 + (size_t)kb * 32 * DIM, hi);
            acc0 = wmma_bf16(af, b0, acc0);
            acc1 = wmma_bf16(af, b1, acc1);
        }
#pragma unroll
        for (int r = 0; r < 8; ++r) {
            int m = r + 8 * hi;
            size_t o = ((size_t)b * KKEEP + kt * 16 + m) * DIM + d0 + nl;
            out[o]      = acc0[r];
            out[o + 16] = acc1[r];
        }
    }
}

// ---------------- launcher ----------------
extern "C" void kernel_launch(void* const* d_in, const int* in_sizes, int n_in,
                              void* d_out, int out_size, void* d_ws, size_t ws_size,
                              hipStream_t stream) {
    const float* xagg   = (const float*)d_in[0];   // hidden_states_for_aggregation [B,N,D]
    const float* xsim   = (const float*)d_in[1];   // hidden_states_for_sim        [B,N,D]
    const float* metric = (const float*)d_in[2];   // cls_attention_map            [B,N]
    float* out = (float*)d_out;                    // [B,K,D] f32 ++ [B,K] int32 (bit pattern)

    // workspace layout (needs ~216 KB)
    float* rinv  = (float*)d_ws;                                   // B*N floats
    int*   bench = (int*)((char*)d_ws + (size_t)BATCH * NTOK * 4); // B*K ints
    int*   high  = bench + BATCH * KKEEP;                          // B*K ints
    int*   out_idx = (int*)out + (size_t)BATCH * KKEEP * DIM;      // tail of d_out

    row_norms  <<<dim3((BATCH * NTOK) / 8), dim3(256), 0, stream>>>(xsim, rinv);
    bench_select<<<dim3(BATCH),             dim3(288), 0, stream>>>(metric, bench, high, out_idx);
    nuwa_main  <<<dim3(KKEEP / 16, BATCH),  dim3(256), 0, stream>>>(xsim, xagg, rinv, bench, high, out);
}
